// GCN_LSTM_Model_65472481460472
// MI455X (gfx1250) — compile-verified, matrix operand order checked by reading, not run on hardware
//
#include <hip/hip_runtime.h>
#include <hip/hip_bf16.h>

// ---------------------------------------------------------------------------
// Problem constants (fixed shapes from the reference)
// ---------------------------------------------------------------------------
#define BB   32          // batch
#define TT   64          // time steps
#define NN   1000        // nodes
#define CC   8           // in channels
#define HG   16          // gcn hidden
#define HL   256         // lstm hidden
#define OUTD 24          // output dim
#define EE   16000       // edges
#define KK   (NN * HG)   // 16000, LSTM input width
#define MM   (BB * TT)   // 2048, GEMM rows
#define GN   (4 * HL)    // 1024, gate width

typedef __attribute__((ext_vector_type(16))) __bf16 bf16x16;
typedef __attribute__((ext_vector_type(8)))  float  f32x8;

union FragU { bf16x16 v; uint4 q[2]; };

__device__ __forceinline__ unsigned short f2bf(float f) {
    unsigned int u = __float_as_uint(f);
    unsigned int r = (u + 0x7FFFu + ((u >> 16) & 1u)) >> 16;  // RNE
    return (unsigned short)r;
}
__device__ __forceinline__ float sigmoidf_(float x) {
    return 1.0f / (1.0f + __expf(-x));
}

// ---------------------------------------------------------------------------
// 0) zero scratch (deg + agg0)
// ---------------------------------------------------------------------------
__global__ void zero_kernel(float* __restrict__ p, int n) {
    for (int i = blockIdx.x * blockDim.x + threadIdx.x; i < n;
         i += gridDim.x * blockDim.x)
        p[i] = 0.0f;
}

// ---------------------------------------------------------------------------
// 1) degree accumulation: deg[col[e]] += w[e]   (consumers add the +1)
// ---------------------------------------------------------------------------
__global__ void deg_kernel(const int* __restrict__ ei,
                           const float* __restrict__ w,
                           float* __restrict__ deg) {
    int e = blockIdx.x * blockDim.x + threadIdx.x;
    if (e < EE) atomicAdd(&deg[ei[EE + e]], w[e]);
}

// ---------------------------------------------------------------------------
// 2) W_ih f32 -> bf16 (packed 4 at a time)
// ---------------------------------------------------------------------------
__global__ void cvt_wih_kernel(const float* __restrict__ src,
                               unsigned short* __restrict__ dst, int n4) {
    for (int i = blockIdx.x * blockDim.x + threadIdx.x; i < n4;
         i += gridDim.x * blockDim.x) {
        float4 f = ((const float4*)src)[i];
        uint2 o;
        o.x = (unsigned)f2bf(f.x) | ((unsigned)f2bf(f.y) << 16);
        o.y = (unsigned)f2bf(f.z) | ((unsigned)f2bf(f.w) << 16);
        ((uint2*)dst)[i] = o;
    }
}

// ---------------------------------------------------------------------------
// 3) edge scatter (batch 0 only): agg0[t,col,:] += norm(e) * (x[0,t,row,:] @ Wg)
// ---------------------------------------------------------------------------
__global__ __launch_bounds__(256)
void scatter_kernel(const float* __restrict__ x,
                    const int* __restrict__ ei,
                    const float* __restrict__ w,
                    const float* __restrict__ Wg,   // [C][HG]
                    const float* __restrict__ deg,
                    float* __restrict__ agg0) {
    __shared__ float wg[CC * HG];
    if (threadIdx.x < CC * HG) wg[threadIdx.x] = Wg[threadIdx.x];
    __syncthreads();

    int idx = blockIdx.x * blockDim.x + threadIdx.x;     // over T*E
    if (idx >= TT * EE) return;
    int t = idx / EE, e = idx - t * EE;
    int r = ei[e], c = ei[EE + e];
    float nrm = rsqrtf(deg[r] + 1.0f) * w[e] * rsqrtf(deg[c] + 1.0f);

    const float4* xp = (const float4*)(x + ((size_t)t * NN + r) * CC);
    float4 x0 = xp[0], x1 = xp[1];
    float* dst = agg0 + ((size_t)t * NN + c) * HG;
#pragma unroll
    for (int h = 0; h < HG; ++h) {
        float s = x0.x * wg[0 * HG + h] + x0.y * wg[1 * HG + h] +
                  x0.z * wg[2 * HG + h] + x0.w * wg[3 * HG + h] +
                  x1.x * wg[4 * HG + h] + x1.y * wg[5 * HG + h] +
                  x1.z * wg[6 * HG + h] + x1.w * wg[7 * HG + h];
        atomicAdd(&dst[h], nrm * s);
    }
}

// ---------------------------------------------------------------------------
// 4) seq[b,t, n*HG+h] = bf16( xw*scale + (b==0 ? agg0 : 0) + b_gcn )
// ---------------------------------------------------------------------------
__global__ __launch_bounds__(256)
void seq_kernel(const float* __restrict__ x,
                const float* __restrict__ Wg,
                const float* __restrict__ bg,
                const float* __restrict__ deg,
                const float* __restrict__ agg0,
                unsigned short* __restrict__ seq) {
    __shared__ float wg[CC * HG];
    __shared__ float bb[HG];
    if (threadIdx.x < CC * HG) wg[threadIdx.x] = Wg[threadIdx.x];
    if (threadIdx.x < HG) bb[threadIdx.x] = bg[threadIdx.x];
    __syncthreads();

    int idx = blockIdx.x * blockDim.x + threadIdx.x;     // over B*T*N
    if (idx >= BB * TT * NN) return;
    int n = idx % NN;
    int bt = idx / NN;
    int b = bt / TT, t = bt % TT;

    const float4* xp = (const float4*)(x + (size_t)idx * CC);
    float4 x0 = xp[0], x1 = xp[1];
    float scale = (b == 0) ? 1.0f / (deg[n] + 1.0f) : 1.0f;
    const float* ag = agg0 + ((size_t)t * NN + n) * HG;
    unsigned short* dst = seq + (size_t)bt * KK + (size_t)n * HG;
#pragma unroll
    for (int h = 0; h < HG; ++h) {
        float s = x0.x * wg[0 * HG + h] + x0.y * wg[1 * HG + h] +
                  x0.z * wg[2 * HG + h] + x0.w * wg[3 * HG + h] +
                  x1.x * wg[4 * HG + h] + x1.y * wg[5 * HG + h] +
                  x1.z * wg[6 * HG + h] + x1.w * wg[7 * HG + h];
        float v = s * scale + (b == 0 ? ag[h] : 0.0f) + bb[h];
        dst[h] = f2bf(v);
    }
}

// ---------------------------------------------------------------------------
// 5) WMMA GEMM: gates_pre[M=2048, GN=1024] = seq(bf16) @ Wih(bf16)^T, f32 acc
//    Block: 256 thr (8 waves), tile 128x64, BK=32, double-buffered LDS.
//    Wave(w): rows [ (w&3)*32 ), cols [ (w>>2)*32 ) -> 4x v_wmma 16x16x32.
// ---------------------------------------------------------------------------
#define Bb 128
#define BN 64
#define BK 32
#define LDT 40   // u16 row stride in LDS (pad keeps 16B alignment, spreads banks)

__global__ __launch_bounds__(256)
void gemm_kernel(const unsigned short* __restrict__ A,   // [MM][KK]
                 const unsigned short* __restrict__ Bm,  // [GN][KK]
                 float* __restrict__ C) {                // [MM][GN]
    __shared__ unsigned short As[2][BB * 0 + Bb * LDT];
    __shared__ unsigned short Bs[2][BN * LDT];

    const int tid   = threadIdx.x;
    const int lane  = tid & 31;
    const int wave  = tid >> 5;
    const int waveM = wave & 3;        // 0..3 -> 32-row strip
    const int waveN = wave >> 2;       // 0..1 -> 32-col strip
    const int mrow  = lane & 15;
    const int khalf = (lane >> 4) * 8; // 0 or 8

    const int m0 = blockIdx.y * Bb;
    const int n0 = blockIdx.x * BN;

    // cooperative-load decomposition: 16B (8 x u16) chunks
    const int aRow0 = (tid) >> 2,        aKc0 = (tid) & 3;
    const int aRow1 = (tid + 256) >> 2,  aKc1 = (tid + 256) & 3;
    const int bRow  = tid >> 2,          bKc  = tid & 3;

    const int NKIT = KK / BK;            // 500

    f32x8 acc[2][2];
#pragma unroll
    for (int i = 0; i < 2; ++i)
#pragma unroll
        for (int j = 0; j < 2; ++j) acc[i][j] = (f32x8){};

    uint4 a0, a1, b0;
    // prologue: global -> regs -> LDS buf 0
    a0 = *(const uint4*)(A + (size_t)(m0 + aRow0) * KK + aKc0 * 8);
    a1 = *(const uint4*)(A + (size_t)(m0 + aRow1) * KK + aKc1 * 8);
    b0 = *(const uint4*)(Bm + (size_t)(n0 + bRow) * KK + bKc * 8);
    *(uint4*)&As[0][aRow0 * LDT + aKc0 * 8] = a0;
    *(uint4*)&As[0][aRow1 * LDT + aKc1 * 8] = a1;
    *(uint4*)&Bs[0][bRow * LDT + bKc * 8]   = b0;
    __syncthreads();

    for (int kt = 0; kt < NKIT; ++kt) {
        const int cur = kt & 1, nxt = cur ^ 1;
        const int kNext = (kt + 1) * BK;
        if (kt + 1 < NKIT) {
            a0 = *(const uint4*)(A + (size_t)(m0 + aRow0) * KK + kNext + aKc0 * 8);
            a1 = *(const uint4*)(A + (size_t)(m0 + aRow1) * KK + kNext + aKc1 * 8);
            b0 = *(const uint4*)(Bm + (size_t)(n0 + bRow) * KK + kNext + bKc * 8);
        }
        if (kt + 2 < NKIT) {  // global_prefetch_b8 two tiles ahead
            __builtin_prefetch(A + (size_t)(m0 + aRow0) * KK + (kt + 2) * BK, 0, 1);
            __builtin_prefetch(Bm + (size_t)(n0 + bRow) * KK + (kt + 2) * BK, 0, 1);
        }

        // fragments from LDS: two ds_load_b128 each (layout-exact for wave32 bf16)
        FragU fa[2], fb[2];
#pragma unroll
        for (int mi = 0; mi < 2; ++mi) {
            int r = (waveM * 32 + mi * 16 + mrow) * LDT;
            fa[mi].q[0] = *(const uint4*)&As[cur][r + khalf];
            fa[mi].q[1] = *(const uint4*)&As[cur][r + 16 + khalf];
        }
#pragma unroll
        for (int ni = 0; ni < 2; ++ni) {
            int r = (waveN * 32 + ni * 16 + mrow) * LDT;
            fb[ni].q[0] = *(const uint4*)&Bs[cur][r + khalf];
            fb[ni].q[1] = *(const uint4*)&Bs[cur][r + 16 + khalf];
        }
#pragma unroll
        for (int mi = 0; mi < 2; ++mi)
#pragma unroll
            for (int ni = 0; ni < 2; ++ni)
                acc[mi][ni] = __builtin_amdgcn_wmma_f32_16x16x32_bf16(
                    false, fa[mi].v, false, fb[ni].v,
                    (short)0, acc[mi][ni], false, false);

        if (kt + 1 < NKIT) {
            *(uint4*)&As[nxt][aRow0 * LDT + aKc0 * 8] = a0;
            *(uint4*)&As[nxt][aRow1 * LDT + aKc1 * 8] = a1;
            *(uint4*)&Bs[nxt][bRow * LDT + bKc * 8]   = b0;
        }
        __syncthreads();
    }

    // C layout: lane L, VGPR e -> M = (L/16)*8 + e, N = L%16
#pragma unroll
    for (int mi = 0; mi < 2; ++mi)
#pragma unroll
        for (int ni = 0; ni < 2; ++ni) {
            int colg = n0 + waveN * 32 + ni * 16 + mrow;
#pragma unroll
            for (int e = 0; e < 8; ++e) {
                int rowg = m0 + waveM * 32 + mi * 16 + khalf + e;
                C[(size_t)rowg * GN + colg] = acc[mi][ni][e];
            }
        }
}

// ---------------------------------------------------------------------------
// 6) LSTM recurrence + projection. One block per batch, thread j owns unit j.
// ---------------------------------------------------------------------------
__global__ __launch_bounds__(256)
void lstm_kernel(const float* __restrict__ gates_pre,  // [MM][GN]
                 const float* __restrict__ Whh,        // [GN][HL]
                 const float* __restrict__ b_ih,
                 const float* __restrict__ b_hh,
                 const float* __restrict__ Wproj,      // [OUTD][HL]
                 const float* __restrict__ b_proj,
                 float* __restrict__ out) {            // [BB][OUTD]
    __shared__ float hs[HL];
    const int b = blockIdx.x, j = threadIdx.x;
    float h = 0.0f, c = 0.0f;

    for (int t = 0; t < TT; ++t) {
        hs[j] = h;
        __syncthreads();
        const float* gp = gates_pre + ((size_t)b * TT + t) * GN;
        float g4[4];
#pragma unroll
        for (int g = 0; g < 4; ++g) {
            const int row = g * HL + j;
            const float4* wr = (const float4*)(Whh + (size_t)row * HL);
            float s = 0.0f;
#pragma unroll 4
            for (int k = 0; k < HL / 4; ++k) {
                float4 w = wr[k];
                s += w.x * hs[4 * k + 0] + w.y * hs[4 * k + 1] +
                     w.z * hs[4 * k + 2] + w.w * hs[4 * k + 3];
            }
            g4[g] = gp[row] + s + b_ih[row] + b_hh[row];
        }
        float ig = sigmoidf_(g4[0]);
        float fg = sigmoidf_(g4[1]);
        float gg = tanhf(g4[2]);
        float og = sigmoidf_(g4[3]);
        c = fg * c + ig * gg;
        h = og * tanhf(c);
        __syncthreads();
    }
    hs[j] = h;
    __syncthreads();
    if (j < OUTD) {
        const float* wr = Wproj + (size_t)j * HL;
        float s = b_proj[j];
        for (int k = 0; k < HL; ++k) s += wr[k] * hs[k];
        out[b * OUTD + j] = s;
    }
}

// ---------------------------------------------------------------------------
// launcher
// ---------------------------------------------------------------------------
extern "C" void kernel_launch(void* const* d_in, const int* in_sizes, int n_in,
                              void* d_out, int out_size, void* d_ws, size_t ws_size,
                              hipStream_t stream) {
    const float* x      = (const float*)d_in[0];
    const int*   ei     = (const int*)d_in[1];
    const float* ew     = (const float*)d_in[2];
    const float* Wg     = (const float*)d_in[3];
    const float* bg     = (const float*)d_in[4];
    const float* Wih    = (const float*)d_in[5];
    const float* Whh    = (const float*)d_in[6];
    const float* bih    = (const float*)d_in[7];
    const float* bhh    = (const float*)d_in[8];
    const float* Wproj  = (const float*)d_in[9];
    const float* bproj  = (const float*)d_in[10];
    float* out          = (float*)d_out;

    char* ws = (char*)d_ws;
    // workspace layout (256B-aligned slabs)
    float*          deg     = (float*)(ws + 0);                    //  4 KB
    float*          agg0    = (float*)(ws + 4096);                 //  4.10 MB
    unsigned short* seq     = (unsigned short*)(ws + 4100096);     // 65.54 MB
    unsigned short* wih_bf  = (unsigned short*)(ws + 69636096);    // 32.77 MB
    float*          gpre    = (float*)(ws + 102404096);            //  8.39 MB
    (void)ws_size; (void)in_sizes; (void)n_in; (void)out_size;

    const int zeroN = (4096 + TT * NN * HG * 4) / 4;   // deg + agg0 floats
    zero_kernel<<<1024, 256, 0, stream>>>(deg, zeroN);

    deg_kernel<<<(EE + 255) / 256, 256, 0, stream>>>(ei, ew, deg);

    cvt_wih_kernel<<<2048, 256, 0, stream>>>(Wih, wih_bf, GN * KK / 4);

    scatter_kernel<<<(TT * EE + 255) / 256, 256, 0, stream>>>(
        x, ei, ew, Wg, deg, agg0);

    seq_kernel<<<(BB * TT * NN + 255) / 256, 256, 0, stream>>>(
        x, Wg, bg, deg, agg0, seq);

    dim3 gGemm(GN / BN, MM / BB * 0 + MM / 128);   // (16, 16)
    gemm_kernel<<<gGemm, 256, 0, stream>>>(seq, wih_bf, gpre);

    lstm_kernel<<<BB, HL, 0, stream>>>(gpre, Whh, bih, bhh, Wproj, bproj, out);
}
